// StructuralBreakDetector_70317204570562
// MI455X (gfx1250) — compile-verified, matrix operand order checked by reading, not run on hardware
//
#include <hip/hip_runtime.h>
#include <hip/hip_bf16.h>
#include <math.h>

typedef __attribute__((ext_vector_type(16))) __bf16 v16bf;
typedef __attribute__((ext_vector_type(8)))  __bf16 v8bf;
typedef __attribute__((ext_vector_type(8)))  float  v8f;

static constexpr int SEQ  = 512;   // T
static constexpr int NB   = 1024;  // batch
static constexpr int NW   = 30;    // wavelets
static constexpr int LW   = 300;   // ricker length
static constexpr int HD   = 64;    // LSTM hidden
static constexpr int FCIN = 14400; // 16*7*128 + 64

__device__ __forceinline__ int koff(int j, int h) {
  return (j < 8) ? (8 * h + j) : (16 + 8 * h + (j - 8));
}

__device__ __forceinline__ v8f wmma_bf16(v16bf a, v16bf b, v8f c) {
  // D = A(16x32 bf16) x B(32x16 bf16) + C(f32)  -> v_wmma_f32_16x16x32_bf16
  return __builtin_amdgcn_wmma_f32_16x16x32_bf16(false, a, false, b, (short)0, c, false, false);
}

__device__ __forceinline__ float sigf(float x) { return 1.0f / (1.0f + __expf(-x)); }

// ---------------- init: ricker bank packed as A-fragments (bf16) -------------
// bankf layout: [mt(2)][kc(10)][lane(32)][16 bf16]
__global__ void k_bank_frags(__bf16* bankf) {
  int blk  = blockIdx.x;            // mt*10 + kc
  int mt   = blk / 10, kc = blk % 10;
  int lane = threadIdx.x;
  int h = lane >> 4, c16 = lane & 15;
  int m = mt * 16 + c16;            // wavelet row (A row = lane%16)
  __bf16* dst = bankf + ((size_t)blk * 32 + lane) * 16;
  for (int j = 0; j < 16; ++j) {
    int k = kc * 32 + koff(j, h);
    float v = 0.f;
    if (m < NW && k < LW) {
      int w  = m + 1;
      int M  = 10 * w;                       // min(10w, 512) = 10w for w<=30
      int s  = 149 - (M - 1) / 2;            // embed offset
      int lo = (LW - 1) - k;                 // flip -> correlation
      int mm = lo - s;
      if (mm >= 0 && mm < M) {
        float x  = (float)mm - (float)(M - 1) * 0.5f;
        float ww = (float)w;
        float A  = 2.0f / (sqrtf(3.0f * ww) * 1.3313353638003897f); // pi^0.25
        float r  = x * x / (ww * ww);
        v = A * (1.0f - r) * __expf(-0.5f * r);
      }
    }
    dst[j] = (__bf16)v;
  }
}

// ---------------- init: W_hh packed as B-fragments (bf16) --------------------
// whhf layout: [nt(16)][kc(2)][lane(32)][16 bf16];  B[k,g] = W_hh[g,k]
__global__ void k_whh_frags(const float* whh, __bf16* out) {
  int blk = blockIdx.x;             // nt*2 + kc
  int lane = threadIdx.x;
  int h = lane >> 4, c16 = lane & 15;
  int nt = blk >> 1, kc = blk & 1;
  int g = nt * 16 + c16;
  __bf16* dst = out + ((size_t)blk * 32 + lane) * 16;
  for (int j = 0; j < 16; ++j) {
    int k = kc * 32 + koff(j, h);
    dst[j] = (__bf16)whh[g * HD + k];
  }
}

// ---------------- init: fc1_w packed as B-fragments (bf16) -------------------
// fc1f layout: [kc(450)][nt(4)][lane(32)][16 bf16];  B[k,n] = fc1_w[n,k]
__global__ void k_fc1_frags(const float* w, __bf16* out) {
  int blk = blockIdx.x;             // kc*4 + nt
  int lane = threadIdx.x;
  int h = lane >> 4, c16 = lane & 15;
  int kc = blk >> 2, nt = blk & 3;
  int g = nt * 16 + c16;
  __bf16* dst = out + ((size_t)blk * 32 + lane) * 16;
  for (int j = 0; j < 16; ++j) {
    int k = kc * 32 + koff(j, h);
    dst[j] = (__bf16)w[(size_t)g * FCIN + k];
  }
}

// ---------------- CWT as WMMA GEMM: out[b,w,t] = sum_l bank[w,l]*xpad[t+l] ---
__global__ void __launch_bounds__(256) k_cwt(const float* spread, const __bf16* bankf,
                                             float* wt) {
  __shared__ __bf16 xp[832];        // xpad[i] = x[b, i-149], i in [0,810]
  int b = blockIdx.x;
  int tid = threadIdx.x;
  for (int i = tid; i < 832; i += 256) {
    int g = i - 149;
    float v = (g >= 0 && g < SEQ) ? spread[(size_t)b * SEQ + g] : 0.f;
    xp[i] = (__bf16)v;
  }
  __syncthreads();
  int wv = tid >> 5, lane = tid & 31;
  int h = lane >> 4, c16 = lane & 15;
  for (int q = 0; q < 4; ++q) {
    int nt = wv * 4 + q;            // 32 t-tiles, 8 waves x 4
    int t0 = nt * 16;
    v8f a0 = {}; v8f a1 = {};
    for (int kc = 0; kc < 10; ++kc) {
      v16bf fa0 = *(const v16bf*)(bankf + ((size_t)(0 * 10 + kc) * 32 + lane) * 16);
      v16bf fa1 = *(const v16bf*)(bankf + ((size_t)(1 * 10 + kc) * 32 + lane) * 16);
      v16bf fb;
      int base = t0 + c16 + kc * 32 + 8 * h;   // B[k,n] = xpad[t0+n+k]
#pragma unroll
      for (int j = 0; j < 8; ++j) fb[j] = xp[base + j];
#pragma unroll
      for (int j = 0; j < 8; ++j) fb[8 + j] = xp[base + 16 + j];
      a0 = wmma_bf16(fa0, fb, a0);
      a1 = wmma_bf16(fa1, fb, a1);
    }
    int t = t0 + c16;
#pragma unroll
    for (int r = 0; r < 8; ++r) {
      int w0 = 8 * h + r;
      wt[((size_t)b * NW + w0) * SEQ + t] = a0[r];
      int w1 = 16 + 8 * h + r;
      if (w1 < NW) wt[((size_t)b * NW + w1) * SEQ + t] = a1[r];
    }
  }
}

// ---------------- conv1 (1->8, 3x3 SAME) + relu + 2x2 maxpool ----------------
__global__ void k_conv1(const float* wt, const float* w, const float* bias, float* p1) {
  size_t idx = (size_t)blockIdx.x * 256 + threadIdx.x;
  if (idx >= (size_t)NB * 8 * 15 * 256) return;
  int ow = (int)(idx % 256); size_t r = idx / 256;
  int oh = (int)(r % 15); r /= 15;
  int oc = (int)(r % 8);  int b = (int)(r / 8);
  float k9[9];
#pragma unroll
  for (int i = 0; i < 9; ++i) k9[i] = w[oc * 9 + i];
  float bv = bias[oc];
  const float* img = wt + (size_t)b * NW * SEQ;
  float mx = 0.f;   // relu floor
#pragma unroll
  for (int py = 0; py < 2; ++py)
#pragma unroll
    for (int px = 0; px < 2; ++px) {
      int ch = 2 * oh + py, cw = 2 * ow + px;
      float s = bv;
#pragma unroll
      for (int dy = -1; dy <= 1; ++dy)
#pragma unroll
        for (int dx = -1; dx <= 1; ++dx) {
          int ih = ch + dy, iw = cw + dx;
          float v = (ih >= 0 && ih < NW && iw >= 0 && iw < SEQ) ? img[ih * SEQ + iw] : 0.f;
          s += v * k9[(dy + 1) * 3 + (dx + 1)];
        }
      mx = fmaxf(mx, fmaxf(s, 0.f));
    }
  p1[idx] = mx;
}

// ---------------- conv2 (8->16) + relu + pool -> comb bf16 (flatten order) ---
__global__ void __launch_bounds__(256) k_conv2(const float* p1, const float* w,
                                               const float* bias, __bf16* comb) {
  __shared__ float wsh[16 * 8 * 9];
  for (int i = threadIdx.x; i < 16 * 8 * 9; i += 256) wsh[i] = w[i];
  __syncthreads();
  size_t idx = (size_t)blockIdx.x * 256 + threadIdx.x;
  if (idx >= (size_t)NB * 16 * 7 * 128) return;
  int ow = (int)(idx % 128); size_t r = idx / 128;
  int oh = (int)(r % 7); r /= 7;
  int oc = (int)(r % 16); int b = (int)(r / 16);
  float cv[4];
  float bv = bias[oc];
  cv[0] = cv[1] = cv[2] = cv[3] = bv;
  for (int ic = 0; ic < 8; ++ic) {
    const float* img = p1 + ((size_t)b * 8 + ic) * 15 * 256;
    const float* kk = wsh + (oc * 8 + ic) * 9;
#pragma unroll
    for (int py = 0; py < 2; ++py)
#pragma unroll
      for (int px = 0; px < 2; ++px) {
        int ch = 2 * oh + py, cw = 2 * ow + px;
        float s = 0.f;
#pragma unroll
        for (int dy = -1; dy <= 1; ++dy)
#pragma unroll
          for (int dx = -1; dx <= 1; ++dx) {
            int ih = ch + dy, iw = cw + dx;
            float v = (ih >= 0 && ih < 15 && iw >= 0 && iw < 256) ? img[ih * 256 + iw] : 0.f;
            s += v * kk[(dy + 1) * 3 + (dx + 1)];
          }
        cv[py * 2 + px] += s;
      }
  }
  float mx = fmaxf(fmaxf(fmaxf(cv[0], 0.f), fmaxf(cv[1], 0.f)),
                   fmaxf(fmaxf(cv[2], 0.f), fmaxf(cv[3], 0.f)));
  comb[(size_t)b * FCIN + oc * 896 + oh * 128 + ow] = (__bf16)mx;
}

// ---------------- LSTM: per-wave 16-batch tile, WMMA gates GEMM each step ----
__global__ void __launch_bounds__(128) k_lstm(const float* spread, const float* yp,
                                              const float* xpr, const float* wih,
                                              const float* bih, const float* bhh,
                                              const __bf16* whhf, __bf16* comb) {
  __shared__ __align__(32) __bf16 whl[16 * 2 * 32 * 16];  // 32 KB W_hh B-frags
  __shared__ float4 wib[256];                              // W_ih rows + bias
  __shared__ __align__(16) __bf16 hbuf[4][16 * 64];        // per-wave h (bf16)
  __shared__ float4 xst[4][16];                            // per-step x stage
  int tid = threadIdx.x;
  {
    const uint4* src = (const uint4*)whhf;
    uint4* dst = (uint4*)whl;
    for (int i = tid; i < (16 * 2 * 32 * 16 * 2) / 16; i += 128) dst[i] = src[i];
  }
  for (int g = tid; g < 256; g += 128)
    wib[g] = make_float4(wih[g * 3 + 0], wih[g * 3 + 1], wih[g * 3 + 2], bih[g] + bhh[g]);
  for (int i = tid; i < 4 * 16 * 64; i += 128) ((__bf16*)hbuf)[i] = (__bf16)0.f;
  __syncthreads();

  int wv = tid >> 5, lane = tid & 31;
  int h = lane >> 4, c16 = lane & 15;
  int b0 = blockIdx.x * 64 + wv * 16;

  float c[4][8];      // cell state: k = j*16 + c16, row m = 8h + r
#pragma unroll
  for (int j = 0; j < 4; ++j)
#pragma unroll
    for (int r = 0; r < 8; ++r) c[j][r] = 0.f;

  v8f acc[16];
  for (int t = 0; t < SEQ; ++t) {
    if (lane < 16) {
      int b = b0 + lane;
      xst[wv][lane] = make_float4(spread[(size_t)b * SEQ + t], yp[(size_t)b * SEQ + t],
                                  xpr[(size_t)b * SEQ + t], 0.f);
    }
    __syncthreads();
    float4 xr[8];
#pragma unroll
    for (int r = 0; r < 8; ++r) xr[r] = xst[wv][8 * h + r];
    // init acc with x-projection + bias (K=3, VALU)
#pragma unroll
    for (int nt = 0; nt < 16; ++nt) {
      float4 wb = wib[nt * 16 + c16];
#pragma unroll
      for (int r = 0; r < 8; ++r)
        acc[nt][r] = wb.w + wb.x * xr[r].x + wb.y * xr[r].y + wb.z * xr[r].z;
    }
    // A fragments of h (row m = c16)
    v16bf a0, a1;
    {
      const v8bf* p = (const v8bf*)&hbuf[wv][c16 * 64 + 8 * h];
      v8bf lo0 = p[0], hi0 = p[2], lo1 = p[4], hi1 = p[6];
#pragma unroll
      for (int j = 0; j < 8; ++j) {
        a0[j] = lo0[j]; a0[8 + j] = hi0[j];
        a1[j] = lo1[j]; a1[8 + j] = hi1[j];
      }
    }
    // gates += h @ W_hh^T : 16 N-tiles x 2 K-chunks
#pragma unroll
    for (int nt = 0; nt < 16; ++nt) {
      v16bf fb0 = *(const v16bf*)(whl + ((nt * 2 + 0) * 32 + lane) * 16);
      acc[nt] = wmma_bf16(a0, fb0, acc[nt]);
      v16bf fb1 = *(const v16bf*)(whl + ((nt * 2 + 1) * 32 + lane) * 16);
      acc[nt] = wmma_bf16(a1, fb1, acc[nt]);
    }
    // activations + state update (gate order i,f,g,o in 4x64 quadrants)
#pragma unroll
    for (int j = 0; j < 4; ++j)
#pragma unroll
      for (int r = 0; r < 8; ++r) {
        float ig = sigf(acc[j][r]);
        float fg = sigf(acc[4 + j][r]);
        float gg = tanhf(acc[8 + j][r]);
        float og = sigf(acc[12 + j][r]);
        float cc = fg * c[j][r] + ig * gg;
        c[j][r] = cc;
        float hv = og * tanhf(cc);
        hbuf[wv][(8 * h + r) * 64 + j * 16 + c16] = (__bf16)hv;
      }
    __syncthreads();
  }
  // final h -> comb[b, 14336 + k]
#pragma unroll
  for (int j = 0; j < 4; ++j)
#pragma unroll
    for (int r = 0; r < 8; ++r) {
      int m = 8 * h + r;
      comb[(size_t)(b0 + m) * FCIN + 14336 + j * 16 + c16] =
          hbuf[wv][m * 64 + j * 16 + c16];
    }
}

// ---------------- FC1: [B,14400] @ [14400,64]^T + bias, relu ----------------
__global__ void __launch_bounds__(128) k_fc1(const __bf16* comb, const __bf16* wf,
                                             const float* bias, float* h1) {
  int tid = threadIdx.x;
  int wv = tid >> 5, lane = tid & 31;
  int h = lane >> 4, c16 = lane & 15;
  int b0 = blockIdx.x * 16;
  int nt = wv;                     // 4 waves cover N=64
  v8f acc = {};
  for (int kc = 0; kc < 450; ++kc) {
    const v8bf* ap = (const v8bf*)(comb + (size_t)(b0 + c16) * FCIN + kc * 32 + 8 * h);
    v8bf alo = ap[0];
    v8bf ahi = ap[2];
    v16bf a;
#pragma unroll
    for (int j = 0; j < 8; ++j) { a[j] = alo[j]; a[8 + j] = ahi[j]; }
    v16bf fb = *(const v16bf*)(wf + ((size_t)(kc * 4 + nt) * 32 + lane) * 16);
    acc = wmma_bf16(a, fb, acc);
  }
  int n = nt * 16 + c16;
  float bv = bias[n];
#pragma unroll
  for (int r = 0; r < 8; ++r) {
    int m = 8 * h + r;
    h1[(size_t)(b0 + m) * 64 + n] = fmaxf(acc[r] + bv, 0.f);
  }
}

// ---------------- FC2 + sigmoid ---------------------------------------------
__global__ void k_fc2(const float* h1, const float* w, const float* b, float* out) {
  int bi = blockIdx.x * 256 + threadIdx.x;
  if (bi >= NB) return;
  float s = b[0];
#pragma unroll 8
  for (int k = 0; k < 64; ++k) s += h1[(size_t)bi * 64 + k] * w[k];
  out[bi] = 1.0f / (1.0f + __expf(-s));
}

extern "C" void kernel_launch(void* const* d_in, const int* in_sizes, int n_in,
                              void* d_out, int out_size, void* d_ws, size_t ws_size,
                              hipStream_t stream) {
  const float* spread = (const float*)d_in[0];
  const float* yprice = (const float*)d_in[1];
  const float* xprice = (const float*)d_in[2];
  const float* c1w = (const float*)d_in[3];
  const float* c1b = (const float*)d_in[4];
  const float* c2w = (const float*)d_in[5];
  const float* c2b = (const float*)d_in[6];
  const float* wih = (const float*)d_in[7];
  const float* whh = (const float*)d_in[8];
  const float* bih = (const float*)d_in[9];
  const float* bhh = (const float*)d_in[10];
  const float* f1w = (const float*)d_in[11];
  const float* f1b = (const float*)d_in[12];
  const float* f2w = (const float*)d_in[13];
  const float* f2b = (const float*)d_in[14];

  char* ws = (char*)d_ws;
  size_t off = 0;
  auto alloc = [&](size_t bytes) {
    size_t p = off;
    off += (bytes + 255) & ~(size_t)255;
    return p;
  };
  __bf16* bankf = (__bf16*)(ws + alloc((size_t)20 * 32 * 16 * 2));
  __bf16* whhf  = (__bf16*)(ws + alloc((size_t)32 * 32 * 16 * 2));
  __bf16* fc1f  = (__bf16*)(ws + alloc((size_t)1800 * 32 * 16 * 2));
  float*  wt    = (float*) (ws + alloc((size_t)NB * NW * SEQ * 4));
  float*  p1    = (float*) (ws + alloc((size_t)NB * 8 * 15 * 256 * 4));
  __bf16* comb  = (__bf16*)(ws + alloc((size_t)NB * FCIN * 2));
  float*  h1    = (float*) (ws + alloc((size_t)NB * 64 * 4));

  k_bank_frags<<<20, 32, 0, stream>>>(bankf);
  k_whh_frags<<<32, 32, 0, stream>>>(whh, whhf);
  k_fc1_frags<<<1800, 32, 0, stream>>>(f1w, fc1f);
  k_cwt<<<NB, 256, 0, stream>>>(spread, bankf, wt);
  k_conv1<<<(int)(((size_t)NB * 8 * 15 * 256 + 255) / 256), 256, 0, stream>>>(wt, c1w, c1b, p1);
  k_conv2<<<(int)(((size_t)NB * 16 * 7 * 128 + 255) / 256), 256, 0, stream>>>(p1, c2w, c2b, comb);
  k_lstm<<<16, 128, 0, stream>>>(spread, yprice, xprice, wih, bih, bhh, whhf, comb);
  k_fc1<<<NB / 16, 128, 0, stream>>>(comb, fc1f, f1b, h1);
  k_fc2<<<(NB + 255) / 256, 256, 0, stream>>>(h1, f2w, f2b, (float*)d_out);
}
